// RotaryAttention_66451734004268
// MI455X (gfx1250) — compile-verified, matrix operand order checked by reading, not run on hardware
//
#include <hip/hip_runtime.h>
#include <hip/hip_bf16.h>

#define D_MODEL 1024
#define N_HEADS 16
#define HEAD_DIM 64
#define SEQ 2048
#define BATCH 4
#define NROW (BATCH * SEQ)   // 8192 flattened (b,s) rows

typedef __attribute__((ext_vector_type(16))) __bf16 v16bf;
typedef __attribute__((ext_vector_type(8)))  float  v8f;

union Frag {
  v16bf  v;
  __bf16 h[16];
  uint4  q[2];
};

__device__ __forceinline__ v8f vzero() {
  v8f z;
#pragma unroll
  for (int i = 0; i < 8; ++i) z[i] = 0.0f;
  return z;
}

// 16x32 bf16 operand fragment, K-contiguous in memory at p + row*ld + k.
// CDNA5 layout (ISA 7.12.2): lanes 0-15 hold K-chunks {0..7,16..23},
// lanes 16-31 hold {8..15,24..31}; vector elems 0..7 = first chunk.
__device__ __forceinline__ v16bf load_frag(const __bf16* __restrict__ p,
                                           int ld, int n0, int k0) {
  const int lane = threadIdx.x & 31;
  const int nn   = lane & 15;
  const int kh   = (lane >> 4) << 3;  // 0 or 8
  const __bf16* base = p + (size_t)(n0 + nn) * ld + k0;
  Frag f;
  f.q[0] = *(const uint4*)(base + kh);
  f.q[1] = *(const uint4*)(base + 16 + kh);
  return f.v;
}

// Same fragment gather, but from an LDS tile (emits ds_load_b128).
__device__ __forceinline__ v16bf frag_lds(const __bf16* base, int ld, int n0) {
  const int lane = threadIdx.x & 31;
  const int nn   = lane & 15;
  const int kh   = (lane >> 4) << 3;
  const __bf16* p = base + (n0 + nn) * ld;
  Frag f;
  f.q[0] = *(const uint4*)(p + kh);
  f.q[1] = *(const uint4*)(p + 16 + kh);
  return f.v;
}

__device__ __forceinline__ v8f wmma_bf16(v16bf a, v16bf b, v8f c) {
  return __builtin_amdgcn_wmma_f32_16x16x32_bf16(false, a, false, b, (short)0, c,
                                                 false, false);
}

// CDNA5 async copy global -> LDS (16B per lane), tracked by ASYNCcnt.
__device__ __forceinline__ void async_copy_b128(uint32_t lds_byte_addr,
                                                const void* gptr) {
  const uint64_t ga = (uint64_t)(uintptr_t)gptr;
  asm volatile("global_load_async_to_lds_b128 %0, %1, off"
               :: "v"(lds_byte_addr), "v"(ga) : "memory");
}

__device__ __forceinline__ uint32_t lds_addr_of(const void* shared_ptr) {
  // LDS aperture: low 32 bits of the flat address are the LDS byte offset.
  return (uint32_t)(uintptr_t)shared_ptr;
}

// ---------------------------------------------------------------- convert
__global__ void cvt_f32_bf16(const float* __restrict__ in,
                             __bf16* __restrict__ out, int n) {
  int i = blockIdx.x * blockDim.x + threadIdx.x;
  if (i < n) out[i] = (__bf16)in[i];
}

// ------------------------------------------------- fused QKV GEMM + RoPE
// grid (NROW/16, 6), block 256 (8 waves). Wave g = by*8+w in [0,48):
// matrix = g/16 (Q,K,V), head = g%16. Each wave computes a 16x64 tile
// (one full head dim) so RoPE partners (idx +/- 32) are tiles t and t^2.
__global__ void __launch_bounds__(256)
qkv_proj_rope(const __bf16* __restrict__ xb,
              const __bf16* __restrict__ Wq, const __bf16* __restrict__ Wk,
              const __bf16* __restrict__ Wv,
              const float* __restrict__ bq, const float* __restrict__ bk,
              const float* __restrict__ bv,
              __bf16* __restrict__ Qo,   // (B,H,S,hd)
              __bf16* __restrict__ Ko,   // (B,H,S,hd)
              __bf16* __restrict__ VTo)  // (B,H,hd,S)  (transposed V)
{
  const int wave = threadIdx.x >> 5;
  const int lane = threadIdx.x & 31;
  const int nn   = lane & 15;
  const int hi   = lane >> 4;
  const int g    = blockIdx.y * 8 + wave;
  const int mat  = g >> 4;
  const int head = g & 15;
  const int row0 = blockIdx.x * 16;

  const __bf16* W    = (mat == 0) ? Wq : (mat == 1) ? Wk : Wv;
  const float*  bias = (mat == 0) ? bq : (mat == 1) ? bk : bv;

  v8f acc[4];
#pragma unroll
  for (int t = 0; t < 4; ++t) acc[t] = vzero();

  const int e0 = head * HEAD_DIM;
  for (int k0 = 0; k0 < D_MODEL; k0 += 32) {
    if (k0 + 32 < D_MODEL)
      __builtin_prefetch(xb + (size_t)(row0 + nn) * D_MODEL + k0 + 32, 0, 1);
    v16bf a = load_frag(xb, D_MODEL, row0, k0);
#pragma unroll
    for (int t = 0; t < 4; ++t) {
      v16bf b = load_frag(W, D_MODEL, e0 + t * 16, k0);
      acc[t] = wmma_bf16(a, b, acc[t]);
    }
  }

  // bias
  float yb[4][8];
#pragma unroll
  for (int t = 0; t < 4; ++t) {
    const float be = bias[e0 + t * 16 + nn];
#pragma unroll
    for (int r = 0; r < 8; ++r) yb[t][r] = acc[t][r] + be;
  }

  if (mat < 2) {  // Q or K: apply RoPE, store (B,H,S,hd)
    __bf16* out = (mat == 0) ? Qo : Ko;
#pragma unroll
    for (int t = 0; t < 4; ++t) {
      const int idx = t * 16 + nn;
      // inv_freq = 10000^{-(idx%32)/32} = exp(-(idx%32) * ln(1e4)/32)
      const float invf = __expf(-(float)(idx & 31) * 0.28782313662f);
#pragma unroll
      for (int r = 0; r < 8; ++r) {
        const int m = row0 + r + (hi << 3);
        const int s = m & (SEQ - 1);
        const int b = m >> 11;
        const float ang = (float)s * invf;
        const float rot = (t < 2) ? -yb[t + 2][r] : yb[t - 2][r];
        const float y = yb[t][r] * __cosf(ang) + rot * __sinf(ang);
        out[(((size_t)b * N_HEADS + head) * SEQ + s) * HEAD_DIM + idx] = (__bf16)y;
      }
    }
  } else {  // V: store transposed (B,H,hd,S) so P.V B-operand is K-contiguous
#pragma unroll
    for (int t = 0; t < 4; ++t) {
      const int idx = t * 16 + nn;
#pragma unroll
      for (int r = 0; r < 8; ++r) {
        const int m = row0 + r + (hi << 3);
        const int s = m & (SEQ - 1);
        const int b = m >> 11;
        VTo[(((size_t)b * N_HEADS + head) * HEAD_DIM + idx) * SEQ + s] =
            (__bf16)yb[t][r];
      }
    }
  }
}

// ----------------------------------------------------- flash attention
// grid 1024, block 256 (8 waves). All 8 waves of a block share one (b,h):
// bh = bx>>4, qTile = (bx&15)*8 + wave. Each iteration the block stages a
// 32-key K tile and the matching V^T tile into LDS with CDNA5 async copies
// (ASYNCcnt), then every wave runs WMMA QK^T + online softmax + WMMA P.V.
__global__ void __launch_bounds__(256)
flash_attn(const __bf16* __restrict__ Q, const __bf16* __restrict__ K,
           const __bf16* __restrict__ VT, __bf16* __restrict__ attn) {
  __shared__ __align__(16) __bf16 sK[32][64];   // 4 KB: K rows kv..kv+31
  __shared__ __align__(16) __bf16 sV[64][32];   // 4 KB: V^T d x 32 keys
  __shared__ float plds[8][16][33];             // per-wave P transpose

  const int tid  = threadIdx.x;
  const int wave = tid >> 5;
  const int lane = tid & 31;
  const int nn   = lane & 15;
  const int hi   = lane >> 4;
  const int bh    = blockIdx.x >> 4;
  const int qTile = (blockIdx.x & 15) * 8 + wave;

  const __bf16* Qb = Q + (size_t)bh * SEQ * HEAD_DIM;
  const __bf16* Kb = K + (size_t)bh * SEQ * HEAD_DIM;
  const __bf16* Vb = VT + (size_t)bh * HEAD_DIM * SEQ;

  // staging assignment: K tile 32x64 (8 chunks/row), V tile 64x32 (4/row)
  const int krow = tid >> 3, kcol = (tid & 7) * 8;
  const int vrow = tid >> 2, vcol = (tid & 3) * 8;
  const uint32_t ldsK = lds_addr_of(&sK[krow][kcol]);
  const uint32_t ldsV = lds_addr_of(&sV[vrow][vcol]);

  const v16bf qa0 = load_frag(Qb, HEAD_DIM, qTile * 16, 0);
  const v16bf qa1 = load_frag(Qb, HEAD_DIM, qTile * 16, 32);

  v8f o[4];
#pragma unroll
  for (int t = 0; t < 4; ++t) o[t] = vzero();
  float mi[8], li[8];
#pragma unroll
  for (int r = 0; r < 8; ++r) { mi[r] = -1e30f; li[r] = 0.0f; }

  for (int kv = 0; kv < SEQ; kv += 32) {
    // cooperative async stage of K/V tiles into LDS
    async_copy_b128(ldsK, Kb + (size_t)(kv + krow) * HEAD_DIM + kcol);
    async_copy_b128(ldsV, Vb + (size_t)vrow * SEQ + kv + vcol);
    asm volatile("s_wait_asynccnt 0" ::: "memory");
    __syncthreads();

    // scores: two 16x16 tiles, K-dim = hd split into 2 WMMAs each
    v16bf kb00 = frag_lds(&sK[0][0], HEAD_DIM, 0);
    v16bf kb01 = frag_lds(&sK[0][32], HEAD_DIM, 0);
    v16bf kb10 = frag_lds(&sK[0][0], HEAD_DIM, 16);
    v16bf kb11 = frag_lds(&sK[0][32], HEAD_DIM, 16);
    v8f s0 = wmma_bf16(qa1, kb01, wmma_bf16(qa0, kb00, vzero()));
    v8f s1 = wmma_bf16(qa1, kb11, wmma_bf16(qa0, kb10, vzero()));
    s0 = s0 * 0.125f;  // hd^-0.5
    s1 = s1 * 0.125f;

    // online softmax per C-layout row (16-lane halves)
#pragma unroll
    for (int r = 0; r < 8; ++r) {
      const float a0 = s0[r], a1 = s1[r];
      float mx = fmaxf(a0, a1);
#pragma unroll
      for (int off = 8; off >= 1; off >>= 1)
        mx = fmaxf(mx, __shfl_xor(mx, off, 32));
      const float mnew  = fmaxf(mi[r], mx);
      const float alpha = __expf(mi[r] - mnew);
      const float p0 = __expf(a0 - mnew);
      const float p1 = __expf(a1 - mnew);
      float rs = p0 + p1;
#pragma unroll
      for (int off = 8; off >= 1; off >>= 1) rs += __shfl_xor(rs, off, 32);
      li[r] = li[r] * alpha + rs;
      mi[r] = mnew;
#pragma unroll
      for (int t = 0; t < 4; ++t) o[t][r] = o[t][r] * alpha;
      const int mrow = r + (hi << 3);
      plds[wave][mrow][nn]      = p0;  // kv col = nn
      plds[wave][mrow][16 + nn] = p1;  // kv col = 16+nn
    }

    // C-layout -> A-layout transpose via LDS (DS ops in-order per wave)
    asm volatile("s_wait_dscnt 0" ::: "memory");
    Frag pf;
    {
      const int mr = nn;
      const int kh = hi << 3;
#pragma unroll
      for (int i = 0; i < 8; ++i) {
        pf.h[i]     = (__bf16)plds[wave][mr][kh + i];
        pf.h[8 + i] = (__bf16)plds[wave][mr][16 + kh + i];
      }
    }

    // O += P (16x32) x V (32x64) from the staged LDS V tile
#pragma unroll
    for (int t = 0; t < 4; ++t) {
      v16bf vb = frag_lds(&sV[0][0], 32, t * 16);
      o[t] = wmma_bf16(pf.v, vb, o[t]);
    }

    __syncthreads();  // WAR: next iteration restages sK/sV
  }

  // normalize + store to (B,S,D) bf16 for the output projection
  const int b = bh >> 4, h = bh & 15;
#pragma unroll
  for (int r = 0; r < 8; ++r) {
    const float inv = 1.0f / li[r];
    const int s = qTile * 16 + r + (hi << 3);
#pragma unroll
    for (int t = 0; t < 4; ++t) {
      const int d = h * HEAD_DIM + t * 16 + nn;
      attn[((size_t)b * SEQ + s) * D_MODEL + d] = (__bf16)(o[t][r] * inv);
    }
  }
}

// ------------------------------------------------------ output projection
__global__ void __launch_bounds__(256)
out_proj(const __bf16* __restrict__ A, const __bf16* __restrict__ Wo,
         const float* __restrict__ bo, float* __restrict__ out) {
  const int wave = threadIdx.x >> 5;
  const int lane = threadIdx.x & 31;
  const int nn   = lane & 15;
  const int hi   = lane >> 4;
  const int cg   = blockIdx.y * 8 + wave;  // column group 0..15
  const int row0 = blockIdx.x * 16;
  const int e0   = cg * 64;

  v8f acc[4];
#pragma unroll
  for (int t = 0; t < 4; ++t) acc[t] = vzero();

  for (int k0 = 0; k0 < D_MODEL; k0 += 32) {
    if (k0 + 32 < D_MODEL)
      __builtin_prefetch(A + (size_t)(row0 + nn) * D_MODEL + k0 + 32, 0, 1);
    v16bf a = load_frag(A, D_MODEL, row0, k0);
#pragma unroll
    for (int t = 0; t < 4; ++t) {
      v16bf b = load_frag(Wo, D_MODEL, e0 + t * 16, k0);
      acc[t] = wmma_bf16(a, b, acc[t]);
    }
  }

#pragma unroll
  for (int t = 0; t < 4; ++t) {
    const int e = e0 + t * 16 + nn;
    const float be = bo[e];
#pragma unroll
    for (int r = 0; r < 8; ++r) {
      const int m = row0 + r + (hi << 3);
      out[(size_t)m * D_MODEL + e] = acc[t][r] + be;
    }
  }
}

// ----------------------------------------------------------------- launch
extern "C" void kernel_launch(void* const* d_in, const int* in_sizes, int n_in,
                              void* d_out, int out_size, void* d_ws,
                              size_t ws_size, hipStream_t stream) {
  const float* x  = (const float*)d_in[0];
  const float* Wq = (const float*)d_in[1];
  const float* bq = (const float*)d_in[2];
  const float* Wk = (const float*)d_in[3];
  const float* bk = (const float*)d_in[4];
  const float* Wv = (const float*)d_in[5];
  const float* bv = (const float*)d_in[6];
  const float* Wo = (const float*)d_in[7];
  const float* bo = (const float*)d_in[8];

  char* w = (char*)d_ws;  // needs ~88 MB
  const size_t MB = (size_t)1 << 20;
  __bf16* xb  = (__bf16*)(w + 0 * MB);    // 16 MB (B*S, D) bf16
  __bf16* Wqb = (__bf16*)(w + 16 * MB);   //  2 MB
  __bf16* Wkb = (__bf16*)(w + 18 * MB);
  __bf16* Wvb = (__bf16*)(w + 20 * MB);
  __bf16* Wob = (__bf16*)(w + 22 * MB);
  __bf16* Qb  = (__bf16*)(w + 24 * MB);   // 16 MB (B,H,S,hd)
  __bf16* Kb  = (__bf16*)(w + 40 * MB);   // 16 MB (B,H,S,hd)
  __bf16* VTb = (__bf16*)(w + 56 * MB);   // 16 MB (B,H,hd,S)
  __bf16* Ab  = (__bf16*)(w + 72 * MB);   // 16 MB (B,S,D)

  const int nX = NROW * D_MODEL;
  const int nW = D_MODEL * D_MODEL;
  cvt_f32_bf16<<<(nX + 255) / 256, 256, 0, stream>>>(x, xb, nX);
  cvt_f32_bf16<<<(nW + 255) / 256, 256, 0, stream>>>(Wq, Wqb, nW);
  cvt_f32_bf16<<<(nW + 255) / 256, 256, 0, stream>>>(Wk, Wkb, nW);
  cvt_f32_bf16<<<(nW + 255) / 256, 256, 0, stream>>>(Wv, Wvb, nW);
  cvt_f32_bf16<<<(nW + 255) / 256, 256, 0, stream>>>(Wo, Wob, nW);

  qkv_proj_rope<<<dim3(NROW / 16, 6), 256, 0, stream>>>(
      xb, Wqb, Wkb, Wvb, bq, bk, bv, Qb, Kb, VTb);

  flash_attn<<<dim3((BATCH * N_HEADS * (SEQ / 16)) / 8), 256, 0, stream>>>(
      Qb, Kb, VTb, Ab);

  out_proj<<<dim3(NROW / 16, 2), 256, 0, stream>>>(Ab, Wob, bo, (float*)d_out);
}
